// PerturbationGenerator_42606075577077
// MI455X (gfx1250) — compile-verified, matrix operand order checked by reading, not run on hardware
//
#include <hip/hip_runtime.h>
#include <math.h>

// ---------------------------------------------------------------------------
// MI455X (gfx1250) fused perturbation-generator.
// Bandwidth budget: hs read once (256MB f32) + fp16 copy written once (128MB)
// ~= 16.5us @ 23.3 TB/s.  Selector GEMM (8.6 GFLOP) runs on
// v_wmma_f32_16x16x32_f16 while streaming; each wave owns a 32-row M tile so
// every B fragment feeds two WMMAs (matrix_b_reuse) and all 8 B loads issue
// in one clause ahead of the WMMA block.
// ---------------------------------------------------------------------------

#define B_  8
#define S_  4096
#define H_  2048
#define DH_ 64
#define K_  256

typedef __attribute__((ext_vector_type(16))) _Float16 v16h;
typedef __attribute__((ext_vector_type(8)))  _Float16 v8h;
typedef __attribute__((ext_vector_type(4)))  float    v4f;
typedef __attribute__((ext_vector_type(8)))  float    v8f;

// ---------------------------------------------------------------------------
// K0: w1 (H x DH, f32, row-major) -> w1t (DH x H, f16) so B-matrix fragments
// are contiguous-in-K v8h loads.  256KB, stays resident in the 192MB L2.
// ---------------------------------------------------------------------------
__global__ void cvt_w1_kernel(const float* __restrict__ w1,
                              _Float16* __restrict__ w1t) {
  int i = blockIdx.x * 256 + threadIdx.x;
  if (i < H_ * DH_) {
    int k = i / DH_;
    int n = i - k * DH_;
    w1t[(size_t)n * H_ + k] = (_Float16)w1[i];
  }
}

// ---------------------------------------------------------------------------
// A-fragment load (ISA 16-bit A 16x32 layout): lane L -> row L&15,
// Kbase=(L>=16)?8:0, VGPR0-3 = K base+0..7, VGPR4-7 = K base+16..23.
// Also emits the fused fp16 copy of hs to the output (non-temporal).
// ---------------------------------------------------------------------------
__device__ __forceinline__ v16h load_a_and_copy(const float* __restrict__ ap,
                                                _Float16* __restrict__ op) {
  v4f f0 = __builtin_nontemporal_load((const v4f*)(ap));
  v4f f1 = __builtin_nontemporal_load((const v4f*)(ap + 4));
  v4f f2 = __builtin_nontemporal_load((const v4f*)(ap + 16));
  v4f f3 = __builtin_nontemporal_load((const v4f*)(ap + 20));
  v16h a;
#pragma unroll
  for (int i = 0; i < 4; ++i) {
    a[i]      = (_Float16)f0[i];
    a[4 + i]  = (_Float16)f1[i];
    a[8 + i]  = (_Float16)f2[i];
    a[12 + i] = (_Float16)f3[i];
  }
  v8h lo, hi;
#pragma unroll
  for (int i = 0; i < 8; ++i) { lo[i] = a[i]; hi[i] = a[8 + i]; }
  __builtin_nontemporal_store(lo, (v8h*)(op));
  __builtin_nontemporal_store(hi, (v8h*)(op + 16));
  return a;
}

__device__ __forceinline__ v16h load_b(const _Float16* __restrict__ bp) {
  v8h x = *(const v8h*)(bp);
  v8h y = *(const v8h*)(bp + 16);
  v16h b;
#pragma unroll
  for (int i = 0; i < 8; ++i) { b[i] = x[i]; b[8 + i] = y[i]; }
  return b;
}

// ---------------------------------------------------------------------------
// K1: fused (hs f32 -> fp16 copy) + selector logits via WMMA.
// 8 waves/block; each wave: 32 rows x H, K-chunks of 32 -> 8 WMMAs/chunk.
// C/D layout: lane holds column n=lane&15; VGPR j -> row j + 8*(lane>>4).
// ---------------------------------------------------------------------------
__global__ __launch_bounds__(256)
void selector_gemm_kernel(const float* __restrict__ hs,
                          const _Float16* __restrict__ w1t,
                          const float* __restrict__ b1,
                          const float* __restrict__ w2,
                          const float* __restrict__ b2,
                          _Float16* __restrict__ outh,
                          float* __restrict__ logits) {
  const int lane = threadIdx.x & 31;
  const int wave = threadIdx.x >> 5;
  const int n    = lane & 15;          // A row / B,C column within tile
  const int kb   = (lane >> 4) << 3;   // 0 or 8
  const size_t row0 = (size_t)blockIdx.x * 256 + (size_t)wave * 32;
  const size_t rowA = row0 + n;        // first 16-row tile
  const size_t rowB = rowA + 16;       // second 16-row tile

  v8f a00, a01, a02, a03;   // accum for rows row0..row0+15, N tiles 0..3
  v8f a10, a11, a12, a13;   // accum for rows row0+16..row0+31
  {
    const float c0 = b1[n], c1 = b1[16 + n], c2 = b1[32 + n], c3 = b1[48 + n];
#pragma unroll
    for (int j = 0; j < 8; ++j) {
      a00[j] = c0; a01[j] = c1; a02[j] = c2; a03[j] = c3;
      a10[j] = c0; a11[j] = c1; a12[j] = c2; a13[j] = c3;
    }
  }

  const float* apA = hs   + rowA * H_ + kb;
  const float* apB = hs   + rowB * H_ + kb;
  _Float16*    opA = outh + rowA * H_ + kb;
  _Float16*    opB = outh + rowB * H_ + kb;
  const _Float16* bp0 = w1t + (size_t)(0 * 16 + n) * H_ + kb;
  const _Float16* bp1 = w1t + (size_t)(1 * 16 + n) * H_ + kb;
  const _Float16* bp2 = w1t + (size_t)(2 * 16 + n) * H_ + kb;
  const _Float16* bp3 = w1t + (size_t)(3 * 16 + n) * H_ + kb;

  for (int k0 = 0; k0 < H_; k0 += 32) {
    // all four B fragments live at once -> one load clause, one wait
    v16h bf0 = load_b(bp0 + k0);
    v16h bf1 = load_b(bp1 + k0);
    v16h bf2 = load_b(bp2 + k0);
    v16h bf3 = load_b(bp3 + k0);

    v16h aA = load_a_and_copy(apA + k0, opA + k0);
    v16h aB = load_a_and_copy(apB + k0, opB + k0);

    // (A0,Bt)(A1,Bt) pairs: second of each pair legitimately reuses B
    a00 = __builtin_amdgcn_wmma_f32_16x16x32_f16(false, aA, false, bf0, (short)0, a00, false, false);
    a10 = __builtin_amdgcn_wmma_f32_16x16x32_f16(false, aB, false, bf0, (short)0, a10, false, true);
    a01 = __builtin_amdgcn_wmma_f32_16x16x32_f16(false, aA, false, bf1, (short)0, a01, false, false);
    a11 = __builtin_amdgcn_wmma_f32_16x16x32_f16(false, aB, false, bf1, (short)0, a11, false, true);
    a02 = __builtin_amdgcn_wmma_f32_16x16x32_f16(false, aA, false, bf2, (short)0, a02, false, false);
    a12 = __builtin_amdgcn_wmma_f32_16x16x32_f16(false, aB, false, bf2, (short)0, a12, false, true);
    a03 = __builtin_amdgcn_wmma_f32_16x16x32_f16(false, aA, false, bf3, (short)0, a03, false, false);
    a13 = __builtin_amdgcn_wmma_f32_16x16x32_f16(false, aB, false, bf3, (short)0, a13, false, true);
  }

  // epilogue: relu + second linear (DH->1) + cross-lane reduce -> logits
  const float w20 = w2[n], w21 = w2[16 + n], w22 = w2[32 + n], w23 = w2[48 + n];
  float p0[8], p1[8];
#pragma unroll
  for (int j = 0; j < 8; ++j) {
    p0[j] = fmaxf(a00[j], 0.f) * w20 + fmaxf(a01[j], 0.f) * w21 +
            fmaxf(a02[j], 0.f) * w22 + fmaxf(a03[j], 0.f) * w23;
    p1[j] = fmaxf(a10[j], 0.f) * w20 + fmaxf(a11[j], 0.f) * w21 +
            fmaxf(a12[j], 0.f) * w22 + fmaxf(a13[j], 0.f) * w23;
  }
  // sum over the 16 columns held by each 16-lane half (xor masks stay in-half)
#pragma unroll
  for (int mask = 1; mask < 16; mask <<= 1) {
#pragma unroll
    for (int j = 0; j < 8; ++j) {
      p0[j] += __shfl_xor(p0[j], mask, 32);
      p1[j] += __shfl_xor(p1[j], mask, 32);
    }
  }
  // lanes 0-15 hold rows 0..7 (index j), lanes 16-31 hold rows 8..15
  if (n < 8) {
    const float bias = b2[0];
    const size_t r = (size_t)((lane >> 4) * 8 + n);
    logits[row0 + r]      = p0[n] + bias;
    logits[row0 + 16 + r] = p1[n] + bias;
  }
}

// ---------------------------------------------------------------------------
// block reductions (wave32)
// ---------------------------------------------------------------------------
__device__ __forceinline__ float waveSum(float v) {
#pragma unroll
  for (int m = 16; m > 0; m >>= 1) v += __shfl_xor(v, m, 32);
  return v;
}
__device__ __forceinline__ float waveMax(float v) {
#pragma unroll
  for (int m = 16; m > 0; m >>= 1) v = fmaxf(v, __shfl_xor(v, m, 32));
  return v;
}
__device__ float blockSum(float v, float* red) {
  __syncthreads();
  v = waveSum(v);
  if ((threadIdx.x & 31) == 0) red[threadIdx.x >> 5] = v;
  __syncthreads();
  float r = (threadIdx.x < (blockDim.x >> 5)) ? red[threadIdx.x] : 0.f;
  if ((threadIdx.x >> 5) == 0) {
    r = waveSum(r);
    if (threadIdx.x == 0) red[0] = r;
  }
  __syncthreads();
  return red[0];
}
__device__ float blockMax(float v, float* red) {
  __syncthreads();
  v = waveMax(v);
  if ((threadIdx.x & 31) == 0) red[threadIdx.x >> 5] = v;
  __syncthreads();
  float r = (threadIdx.x < (blockDim.x >> 5)) ? red[threadIdx.x] : -INFINITY;
  if ((threadIdx.x >> 5) == 0) {
    r = waveMax(r);
    if (threadIdx.x == 0) red[0] = r;
  }
  __syncthreads();
  return red[0];
}

// ---------------------------------------------------------------------------
// K2: per batch row: logsumexp over S, gumbel keys, bitonic top-K (full sort
// of 4096 key/index pairs in LDS, index tiebreak matches jax top_k), outputs.
// ---------------------------------------------------------------------------
__global__ __launch_bounds__(1024)
void topk_kernel(const float* __restrict__ logits,
                 const float* __restrict__ noise_u,
                 int* __restrict__ out_idx,
                 int* __restrict__ out_types,
                 float* __restrict__ out_logprob) {
  __shared__ float skey[S_];
  __shared__ int   sidx[S_];
  __shared__ float red[32];

  const int b   = blockIdx.x;
  const int tid = threadIdx.x;
  const float* lg = logits  + (size_t)b * S_;
  const float* nu = noise_u + (size_t)b * S_;

  // log-softmax stats
  float lmax = -INFINITY;
  for (int s = tid; s < S_; s += 1024) lmax = fmaxf(lmax, lg[s]);
  lmax = blockMax(lmax, red);

  float lsum = 0.f;
  for (int s = tid; s < S_; s += 1024) lsum += expf(lg[s] - lmax);
  lsum = blockSum(lsum, red);
  const float lse = lmax + logf(lsum);

  // gumbel-perturbed keys
  for (int s = tid; s < S_; s += 1024) {
    float g = -logf(-logf(nu[s]));
    skey[s] = lg[s] + g;
    sidx[s] = s;
  }
  __syncthreads();

  // bitonic sort, descending, smaller index first on ties
  for (int k = 2; k <= S_; k <<= 1) {
    for (int j = k >> 1; j > 0; j >>= 1) {
      for (int i = tid; i < S_; i += 1024) {
        int ixj = i ^ j;
        if (ixj > i) {
          float ki = skey[i], kj = skey[ixj];
          int   ii = sidx[i], ij = sidx[ixj];
          bool desc = ((i & k) == 0);
          bool wrongDesc = (ki < kj) || (ki == kj && ii > ij);
          if (wrongDesc == desc) {
            skey[i] = kj; skey[ixj] = ki;
            sidx[i] = ij; sidx[ixj] = ii;
          }
        }
      }
      __syncthreads();
    }
  }

  float lp = 0.f;
  if (tid < K_) {
    int s = sidx[tid];
    out_idx[b * K_ + tid]   = s;
    out_types[b * K_ + tid] = 1;               // FIXED_TYPE
    lp = lg[s] - lse;
  }
  lp = blockSum(lp, red);
  if (tid == 0) out_logprob[b] = lp * (1.0f / (float)K_);
}

// ---------------------------------------------------------------------------
// K3: overwrite selected rows of the fp16 output with the perturb value
// (TYPE_VALUES[1]*SCALE == 0.0).  One block per (b,k): 256 thr x v8h store.
// ---------------------------------------------------------------------------
__global__ __launch_bounds__(256)
void scatter_kernel(const int* __restrict__ sel, _Float16* __restrict__ outh) {
  const int b = blockIdx.x / K_;
  const int k = blockIdx.x - b * K_;
  int s = sel[b * K_ + k];
  s = (s < 0) ? 0 : (s > S_ - 1 ? S_ - 1 : s);
  v8h z;
#pragma unroll
  for (int i = 0; i < 8; ++i) z[i] = (_Float16)0.0f;
  _Float16* row = outh + ((size_t)b * S_ + (size_t)s) * H_;
  __builtin_nontemporal_store(z, (v8h*)(row + threadIdx.x * 8));
}

// ---------------------------------------------------------------------------
extern "C" void kernel_launch(void* const* d_in, const int* in_sizes, int n_in,
                              void* d_out, int out_size, void* d_ws, size_t ws_size,
                              hipStream_t stream) {
  const float* hs      = (const float*)d_in[0];   // (B,S,H) f32
  const float* noise_u = (const float*)d_in[1];   // (B,S)   f32
  const float* w1      = (const float*)d_in[2];   // (H,DH)  f32
  const float* b1      = (const float*)d_in[3];   // (DH,)   f32
  const float* w2      = (const float*)d_in[4];   // (DH,1)  f32
  const float* b2      = (const float*)d_in[5];   // (1,)    f32

  // workspace: w1t f16 (256KB) | logits f32 (128KB)
  _Float16* w1t    = (_Float16*)d_ws;
  float*    logits = (float*)((char*)d_ws + (size_t)DH_ * H_ * sizeof(_Float16));

  // outputs concatenated byte-wise in return order
  char* base = (char*)d_out;
  _Float16* out_perturbed = (_Float16*)base;                       // (B,S,H) f16
  size_t off = (size_t)B_ * S_ * H_ * sizeof(_Float16);
  int* out_idx   = (int*)(base + off);  off += (size_t)B_ * K_ * sizeof(int);
  int* out_types = (int*)(base + off);  off += (size_t)B_ * K_ * sizeof(int);
  float* out_lp  = (float*)(base + off);

  cvt_w1_kernel<<<(H_ * DH_ + 255) / 256, 256, 0, stream>>>(w1, w1t);
  selector_gemm_kernel<<<(B_ * S_) / 256, 256, 0, stream>>>(
      hs, w1t, b1, w2, b2, out_perturbed, logits);
  topk_kernel<<<B_, 1024, 0, stream>>>(logits, noise_u, out_idx, out_types, out_lp);
  scatter_kernel<<<B_ * K_, 256, 0, stream>>>(out_idx, out_perturbed);
}
